// GAMDNet_21809843929776
// MI455X (gfx1250) — matrix operand order repaired and checked by reading, not compiled, not run on hardware
//
#include <hip/hip_runtime.h>
#include <hip/hip_bf16.h>

#define EMBED   128
#define HIDDEN  256
#define NLAYERS 4
#define NNODES  20000
#define NEDGES  640000

typedef __bf16 bf16;
typedef __attribute__((ext_vector_type(16))) __bf16 v16bf;
typedef __attribute__((ext_vector_type(8)))  float  v8f;

// ---------------------------------------------------------------- helpers
// Branch-free tanh-form GELU: outputs are quantized to bf16 (~4e-3 rel), so
// the ~1e-3 error of the tanh form is below the storage noise floor, and it
// avoids libm erff's divergent EXEC-mask branches between WMMAs.
__device__ __forceinline__ float gelu_f(float x) {
    float u = 0.7978845608028654f * (x + 0.044715f * x * x * x);
    float t = 1.0f - 2.0f / (__expf(2.0f * u) + 1.0f);   // tanh(u), saturates
    return 0.5f * x * (1.0f + t);
}
__device__ __forceinline__ float silu_f(float x) {
    return x / (1.0f + __expf(-x));
}

__device__ __forceinline__ v8f wmma_bf16(v16bf a, v16bf b, v8f c) {
    // D = A(16x32 bf16) x B(32x16 bf16) + C(f32)
    return __builtin_amdgcn_wmma_f32_16x16x32_bf16(
        false, a, false, b, (short)0, c, false, false);
}

// A fragment: row-major [16][ldA] bf16 tile, window starting at column k0
// (pass A + k0). Per ISA: lanes 0-15 row M=lane, regs v0..3 K=0..7,
// v4..7 K=16..23; lanes 16-31 same rows, K offset +8.
__device__ __forceinline__ v16bf load_a_frag(const bf16* A, int ldA) {
    int lane = threadIdx.x & 31;
    int row  = lane & 15;
    int half = lane >> 4;
    const bf16* rp = A + row * ldA + (half ? 8 : 0);
    union { v16bf v; unsigned int u[8]; } fr;
#pragma unroll
    for (int v = 0; v < 8; ++v) {
        int k = 2 * (v & 3) + ((v & 4) ? 16 : 0);
        fr.u[v] = *(const unsigned int*)(rp + k);   // K pair (k,k+1)
    }
    return fr.v;
}

// B fragment from column-major (transposed) weights BT[N][K]:
// lane col = lane&15 (+col0); same K layout as A: regs v0..3 K=0..7,
// v4..7 K=16..23; lanes 16-31 K offset +8.
__device__ __forceinline__ v16bf load_b_frag(const bf16* BT, int K, int k0, int col0) {
    int lane  = threadIdx.x & 31;
    int col   = (lane & 15) + col0;
    int khalf = lane >> 4;
    const bf16* cp = BT + (size_t)col * K + k0 + khalf * 8;
    union { v16bf v; unsigned int u[8]; } fr;
#pragma unroll
    for (int v = 0; v < 8; ++v)
        fr.u[v] = *(const unsigned int*)(cp + 2 * (v & 3) + ((v & 4) ? 16 : 0));
    return fr.v;
}

// ------------------------------------------------- weight convert+transpose
// src: row-major f32 [K][N]  ->  dst: col-major bf16 [N][K]
__global__ void k_convert_w(const float* __restrict__ w, bf16* __restrict__ wt,
                            int K, int N) {
    int i = blockIdx.x * blockDim.x + threadIdx.x;
    if (i >= K * N) return;
    int k = i / N, n = i % N;
    wt[(size_t)n * K + k] = (bf16)w[i];
}

// ------------------------------------------------------------ edge embedding
// One wave per 16 edges. MLP 3->256 (VALU) ->256 (WMMA) ->128 (WMMA) + LN.
__global__ __launch_bounds__(32) void k_edge_embed(
    const float* __restrict__ pos, const int* __restrict__ ci,
    const int* __restrict__ ni,
    const float* __restrict__ w0, const float* __restrict__ b0,
    const bf16*  __restrict__ w1t, const float* __restrict__ b1,
    const bf16*  __restrict__ w2t, const float* __restrict__ b2,
    const float* __restrict__ ln_g, const float* __restrict__ ln_b,
    bf16* __restrict__ edge_emb)
{
    __shared__ __align__(16) float s_rel[16][3];
    __shared__ __align__(16) bf16  sA[16][HIDDEN];
    __shared__ __align__(16) bf16  sB[16][HIDDEN];
    __shared__ __align__(16) float sE[16][EMBED];
    __shared__ float s_mu[16], s_rs[16];

    int lane = threadIdx.x;
    int eb   = blockIdx.x * 16;

    if (lane < 16) {
        int e = eb + lane;
        int c = ci[e], n = ni[e];
        s_rel[lane][0] = pos[(size_t)n*3+0] - pos[(size_t)c*3+0];
        s_rel[lane][1] = pos[(size_t)n*3+1] - pos[(size_t)c*3+1];
        s_rel[lane][2] = pos[(size_t)n*3+2] - pos[(size_t)c*3+2];
    }
    __syncthreads();

    // layer0: 3 -> 256, K too small for WMMA, do on VALU
    for (int i = lane; i < 16 * HIDDEN; i += 32) {
        int r = i >> 8, n = i & 255;
        float v = s_rel[r][0] * w0[n] + s_rel[r][1] * w0[256 + n]
                + s_rel[r][2] * w0[512 + n] + b0[n];
        sA[r][n] = (bf16)gelu_f(v);
    }
    __syncthreads();

    int col = lane & 15, half = lane >> 4;

    // layer1: 256 -> 256 (16 N-tiles x 8 K-steps)
    for (int nt = 0; nt < 16; ++nt) {
        v8f acc = {};
#pragma unroll
        for (int kt = 0; kt < 8; ++kt) {
            v16bf a = load_a_frag(&sA[0][kt * 32], HIDDEN);
            v16bf b = load_b_frag(w1t, HIDDEN, kt * 32, nt * 16);
            acc = wmma_bf16(a, b, acc);
        }
        int n = nt * 16 + col;
        float bias = b1[n];
#pragma unroll
        for (int r = 0; r < 8; ++r) {
            int m = r + half * 8;
            sB[m][n] = (bf16)gelu_f(acc[r] + bias);
        }
    }
    __syncthreads();

    // layer2: 256 -> 128 (8 N-tiles x 8 K-steps)
    for (int nt = 0; nt < 8; ++nt) {
        v8f acc = {};
#pragma unroll
        for (int kt = 0; kt < 8; ++kt) {
            v16bf a = load_a_frag(&sB[0][kt * 32], HIDDEN);
            v16bf b = load_b_frag(w2t, HIDDEN, kt * 32, nt * 16);
            acc = wmma_bf16(a, b, acc);
        }
        int n = nt * 16 + col;
        float bias = b2[n];
#pragma unroll
        for (int r = 0; r < 8; ++r) {
            int m = r + half * 8;
            sE[m][n] = acc[r] + bias;
        }
    }
    __syncthreads();

    // LayerNorm over 128 features (f32)
    if (lane < 16) {
        float s = 0.f, ss = 0.f;
        for (int n = 0; n < EMBED; ++n) { float v = sE[lane][n]; s += v; ss += v * v; }
        float mu  = s * (1.0f / EMBED);
        float var = ss * (1.0f / EMBED) - mu * mu;
        s_mu[lane] = mu;
        s_rs[lane] = rsqrtf(var + 1e-5f);
    }
    __syncthreads();
    for (int i = lane; i < 16 * EMBED; i += 32) {
        int r = i >> 7, n = i & 127;
        float v = (sE[r][n] - s_mu[r]) * s_rs[r] * ln_g[n] + ln_b[n];
        edge_emb[(size_t)(eb + r) * EMBED + n] = (bf16)v;
    }
}

// ------------------------------------------------------------- node buffers
__global__ void k_init_node(const float* __restrict__ node_embed,
                            float* __restrict__ node) {
    int i = blockIdx.x * blockDim.x + threadIdx.x;
    if (i < NNODES * EMBED) node[i] = node_embed[i & (EMBED - 1)];
}

// LN(node) -> h_f32 + h_bf16; zero agg. One wave per node row.
__global__ __launch_bounds__(256) void k_node_pre(
    const float* __restrict__ node,
    const float* __restrict__ g, const float* __restrict__ b,
    float* __restrict__ h_f32, bf16* __restrict__ h_bf16,
    float* __restrict__ agg)
{
    int wave = threadIdx.x >> 5, lane = threadIdx.x & 31;
    int row  = blockIdx.x * 8 + wave;
    if (row >= NNODES) return;
    const float* x = node + (size_t)row * EMBED;
    float s = 0.f, ss = 0.f;
#pragma unroll
    for (int n = lane; n < EMBED; n += 32) { float v = x[n]; s += v; ss += v * v; }
#pragma unroll
    for (int off = 16; off > 0; off >>= 1) {
        s  += __shfl_xor(s, off);
        ss += __shfl_xor(ss, off);
    }
    float mu  = s * (1.0f / EMBED);
    float var = ss * (1.0f / EMBED) - mu * mu;
    float rs  = rsqrtf(var + 1e-5f);
#pragma unroll
    for (int n = lane; n < EMBED; n += 32) {
        float v = (x[n] - mu) * rs * g[n] + b[n];
        size_t idx = (size_t)row * EMBED + n;
        h_f32[idx]  = v;
        h_bf16[idx] = (bf16)v;
        agg[idx]    = 0.f;
    }
}

// ------------------------------------------------- per-layer edge messages
// One wave per 16 edges: s = edge_emb + h[ci] + h[ni];
// t = silu(silu(s) @ phi_w0 + b0) @ phi_w1 + b1; scatter-add h[ni]*t into agg.
__global__ __launch_bounds__(32) void k_edge_msg(
    const int* __restrict__ ci, const int* __restrict__ ni,
    const bf16* __restrict__ edge_emb, const bf16* __restrict__ h_bf16,
    const bf16* __restrict__ w0t, const float* __restrict__ b0,
    const bf16* __restrict__ w1t, const float* __restrict__ b1,
    float* __restrict__ agg)
{
    __shared__ int  s_ci[16], s_ni[16];
    __shared__ __align__(16) bf16 sS[16][EMBED];    // silu(s)
    __shared__ __align__(16) bf16 sNb[16][EMBED];   // neighbor features
    __shared__ __align__(16) bf16 sT[16][HIDDEN];   // hidden
    int lane = threadIdx.x;
    int eb   = blockIdx.x * 16;
    if (lane < 16) { s_ci[lane] = ci[eb + lane]; s_ni[lane] = ni[eb + lane]; }
    __syncthreads();

    for (int i = lane; i < 16 * EMBED; i += 32) {
        int r = i >> 7, n = i & 127;
        bf16 nb = h_bf16[(size_t)s_ni[r] * EMBED + n];
        float v = (float)edge_emb[(size_t)(eb + r) * EMBED + n]
                + (float)h_bf16[(size_t)s_ci[r] * EMBED + n] + (float)nb;
        sNb[r][n] = nb;
        sS[r][n]  = (bf16)silu_f(v);
    }
    __syncthreads();

    int col = lane & 15, half = lane >> 4;

    // GEMM1: [16x128] @ [128x256]
    for (int nt = 0; nt < 16; ++nt) {
        v8f acc = {};
#pragma unroll
        for (int kt = 0; kt < 4; ++kt) {
            v16bf a = load_a_frag(&sS[0][kt * 32], EMBED);
            v16bf b = load_b_frag(w0t, EMBED, kt * 32, nt * 16);
            acc = wmma_bf16(a, b, acc);
        }
        int n = nt * 16 + col;
        float bias = b0[n];
#pragma unroll
        for (int r = 0; r < 8; ++r) {
            int m = r + half * 8;
            sT[m][n] = (bf16)silu_f(acc[r] + bias);
        }
    }
    __syncthreads();

    // GEMM2: [16x256] @ [256x128]; msg = nb * t; atomic scatter-add (L2)
    for (int nt = 0; nt < 8; ++nt) {
        v8f acc = {};
#pragma unroll
        for (int kt = 0; kt < 8; ++kt) {
            v16bf a = load_a_frag(&sT[0][kt * 32], HIDDEN);
            v16bf b = load_b_frag(w1t, HIDDEN, kt * 32, nt * 16);
            acc = wmma_bf16(a, b, acc);
        }
        int n = nt * 16 + col;
        float bias = b1[n];
#pragma unroll
        for (int r = 0; r < 8; ++r) {
            int m = r + half * 8;
            float msg = (float)sNb[m][n] * (acc[r] + bias);
            unsafeAtomicAdd(&agg[(size_t)s_ci[m] * EMBED + n], msg);
        }
    }
}

// node = silu(h + agg) @ theta_w + theta_b + node   (one wave per 16 nodes)
__global__ __launch_bounds__(32) void k_node_update(
    const float* __restrict__ h_f32, const float* __restrict__ agg,
    const bf16* __restrict__ thetat, const float* __restrict__ theta_b,
    float* __restrict__ node)
{
    __shared__ __align__(16) bf16 sU[16][EMBED];
    int lane = threadIdx.x;
    int nb0  = blockIdx.x * 16;
    for (int i = lane; i < 16 * EMBED; i += 32) {
        int r = i >> 7, n = i & 127;
        size_t idx = (size_t)(nb0 + r) * EMBED + n;
        sU[r][n] = (bf16)silu_f(h_f32[idx] + agg[idx]);
    }
    __syncthreads();
    int col = lane & 15, half = lane >> 4;
    for (int nt = 0; nt < 8; ++nt) {
        v8f acc = {};
#pragma unroll
        for (int kt = 0; kt < 4; ++kt) {
            v16bf a = load_a_frag(&sU[0][kt * 32], EMBED);
            v16bf b = load_b_frag(thetat, EMBED, kt * 32, nt * 16);
            acc = wmma_bf16(a, b, acc);
        }
        int n = nt * 16 + col;
        float bias = theta_b[n];
#pragma unroll
        for (int r = 0; r < 8; ++r) {
            int m = r + half * 8;
            size_t idx = (size_t)(nb0 + m) * EMBED + n;
            node[idx] = node[idx] + acc[r] + bias;   // residual
        }
    }
}

// ------------------------------------------------------------- force head
__global__ __launch_bounds__(32) void k_force(
    const float* __restrict__ node,
    const bf16* __restrict__ fpw0t, const float* __restrict__ fpb0,
    const float* __restrict__ fpw1, const float* __restrict__ fpb1,
    float* __restrict__ out)
{
    __shared__ __align__(16) bf16 sA[16][EMBED];
    __shared__ __align__(16) bf16 sG[16][HIDDEN];
    int lane = threadIdx.x;
    int nb0  = blockIdx.x * 16;
    for (int i = lane; i < 16 * EMBED; i += 32) {
        int r = i >> 7, n = i & 127;
        sA[r][n] = (bf16)node[(size_t)(nb0 + r) * EMBED + n];
    }
    __syncthreads();
    int col = lane & 15, half = lane >> 4;
    for (int nt = 0; nt < 16; ++nt) {
        v8f acc = {};
#pragma unroll
        for (int kt = 0; kt < 4; ++kt) {
            v16bf a = load_a_frag(&sA[0][kt * 32], EMBED);
            v16bf b = load_b_frag(fpw0t, EMBED, kt * 32, nt * 16);
            acc = wmma_bf16(a, b, acc);
        }
        int n = nt * 16 + col;
        float bias = fpb0[n];
#pragma unroll
        for (int r = 0; r < 8; ++r) {
            int m = r + half * 8;
            sG[m][n] = (bf16)gelu_f(acc[r] + bias);
        }
    }
    __syncthreads();
    // 256 -> 3 on VALU (48 dot products per tile)
    for (int i = lane; i < 48; i += 32) {
        int r = i / 3, d = i % 3;
        float s = fpb1[d];
        for (int k = 0; k < HIDDEN; ++k)
            s += (float)sG[r][k] * fpw1[k * 3 + d];
        out[(size_t)(nb0 + r) * 3 + d] = s;
    }
}

// =========================================================== host launcher
struct P {
    const float *pos, *ee_w0, *ee_b0, *ee_w1, *ee_b1, *ee_w2, *ee_b2,
                *ee_ln_g, *ee_ln_b, *node_embed, *fp_w0, *fp_b0, *fp_w1, *fp_b1;
    const int* eidx;
    const float *ln_g[NLAYERS], *ln_b[NLAYERS], *phi_w0[NLAYERS], *phi_b0[NLAYERS],
                *phi_w1[NLAYERS], *phi_b1[NLAYERS], *theta_w[NLAYERS], *theta_b[NLAYERS];
};

extern "C" void kernel_launch(void* const* d_in, const int* in_sizes, int n_in,
                              void* d_out, int out_size, void* d_ws, size_t ws_size,
                              hipStream_t stream)
{
    (void)n_in; (void)out_size; (void)ws_size;
    P p;
    const float* const* F = (const float* const*)d_in;

    if (in_sizes[0] == 2 * NEDGES) {
        // jax tree-flatten order (dict keys sorted)
        p.eidx = (const int*)d_in[0];
        p.ee_b0 = F[1];  p.ee_b1 = F[2];  p.ee_b2 = F[3];
        p.ee_ln_b = F[4]; p.ee_ln_g = F[5];
        p.ee_w0 = F[6];  p.ee_w1 = F[7];  p.ee_w2 = F[8];
        p.fp_b0 = F[9];  p.fp_b1 = F[10]; p.fp_w0 = F[11]; p.fp_w1 = F[12];
        for (int l = 0; l < NLAYERS; ++l) {
            int b = 13 + 8 * l;   // ln_b, ln_g, phi_b0, phi_b1, phi_w0, phi_w1, theta_b, theta_w
            p.ln_b[l] = F[b + 0]; p.ln_g[l] = F[b + 1];
            p.phi_b0[l] = F[b + 2]; p.phi_b1[l] = F[b + 3];
            p.phi_w0[l] = F[b + 4]; p.phi_w1[l] = F[b + 5];
            p.theta_b[l] = F[b + 6]; p.theta_w[l] = F[b + 7];
        }
        p.node_embed = F[45];
        p.pos = F[46];
    } else {
        // python dict insertion order
        p.pos = F[0];
        p.eidx = (const int*)d_in[1];
        p.ee_w0 = F[2];  p.ee_b0 = F[3];  p.ee_w1 = F[4];  p.ee_b1 = F[5];
        p.ee_w2 = F[6];  p.ee_b2 = F[7];  p.ee_ln_g = F[8]; p.ee_ln_b = F[9];
        for (int l = 0; l < NLAYERS; ++l) {
            int b = 10 + 8 * l;   // ln_g, ln_b, phi_w0, phi_b0, phi_w1, phi_b1, theta_w, theta_b
            p.ln_g[l] = F[b + 0]; p.ln_b[l] = F[b + 1];
            p.phi_w0[l] = F[b + 2]; p.phi_b0[l] = F[b + 3];
            p.phi_w1[l] = F[b + 4]; p.phi_b1[l] = F[b + 5];
            p.theta_w[l] = F[b + 6]; p.theta_b[l] = F[b + 7];
        }
        p.node_embed = F[42];
        p.fp_w0 = F[43]; p.fp_b0 = F[44]; p.fp_w1 = F[45]; p.fp_b1 = F[46];
    }

    const int* ci = p.eidx;
    const int* ni = p.eidx + NEDGES;

    // ---- workspace bump allocator (256B aligned)
    char*  base = (char*)d_ws;
    size_t off  = 0;
    auto alloc = [&](size_t bytes) {
        void* r = base + off;
        off = (off + bytes + 255) & ~(size_t)255;
        return r;
    };
    bf16* w_ee1  = (bf16*)alloc((size_t)HIDDEN * HIDDEN * 2);
    bf16* w_ee2  = (bf16*)alloc((size_t)HIDDEN * EMBED  * 2);
    bf16* w_fp0  = (bf16*)alloc((size_t)EMBED  * HIDDEN * 2);
    bf16* w_phi0[NLAYERS]; bf16* w_phi1[NLAYERS]; bf16* w_theta[NLAYERS];
    for (int l = 0; l < NLAYERS; ++l) {
        w_phi0[l]  = (bf16*)alloc((size_t)EMBED  * HIDDEN * 2);
        w_phi1[l]  = (bf16*)alloc((size_t)HIDDEN * EMBED  * 2);
        w_theta[l] = (bf16*)alloc((size_t)EMBED  * EMBED  * 2);
    }
    bf16*  edge_emb = (bf16*) alloc((size_t)NEDGES * EMBED * 2);
    float* node     = (float*)alloc((size_t)NNODES * EMBED * 4);
    float* h_f32    = (float*)alloc((size_t)NNODES * EMBED * 4);
    bf16*  h_bf16   = (bf16*) alloc((size_t)NNODES * EMBED * 2);
    float* agg      = (float*)alloc((size_t)NNODES * EMBED * 4);

    // ---- convert + transpose weights to bf16 col-major
    auto conv = [&](const float* w, bf16* dst, int K, int N) {
        int tot = K * N;
        k_convert_w<<<(tot + 255) / 256, 256, 0, stream>>>(w, dst, K, N);
    };
    conv(p.ee_w1, w_ee1, HIDDEN, HIDDEN);
    conv(p.ee_w2, w_ee2, HIDDEN, EMBED);
    conv(p.fp_w0, w_fp0, EMBED, HIDDEN);
    for (int l = 0; l < NLAYERS; ++l) {
        conv(p.phi_w0[l],  w_phi0[l],  EMBED,  HIDDEN);
        conv(p.phi_w1[l],  w_phi1[l],  HIDDEN, EMBED);
        conv(p.theta_w[l], w_theta[l], EMBED,  EMBED);
    }

    // ---- edge embedding MLP + LN
    k_edge_embed<<<NEDGES / 16, 32, 0, stream>>>(
        p.pos, ci, ni, p.ee_w0, p.ee_b0, w_ee1, p.ee_b1, w_ee2, p.ee_b2,
        p.ee_ln_g, p.ee_ln_b, edge_emb);

    // ---- node init (broadcast learned embedding)
    k_init_node<<<(NNODES * EMBED + 255) / 256, 256, 0, stream>>>(p.node_embed, node);

    // ---- message-passing layers
    for (int l = 0; l < NLAYERS; ++l) {
        k_node_pre<<<NNODES / 8, 256, 0, stream>>>(
            node, p.ln_g[l], p.ln_b[l], h_f32, h_bf16, agg);
        k_edge_msg<<<NEDGES / 16, 32, 0, stream>>>(
            ci, ni, edge_emb, h_bf16, w_phi0[l], p.phi_b0[l],
            w_phi1[l], p.phi_b1[l], agg);
        k_node_update<<<NNODES / 16, 32, 0, stream>>>(
            h_f32, agg, w_theta[l], p.theta_b[l], node);
    }

    // ---- force prediction head
    k_force<<<NNODES / 16, 32, 0, stream>>>(
        node, w_fp0, p.fp_b0, p.fp_w1, p.fp_b1, (float*)d_out);
}
